// Model_14259291422795
// MI455X (gfx1250) — matrix operand order repaired and checked by reading, not compiled, not run on hardware
//
#include <hip/hip_runtime.h>
#include <hip/hip_bf16.h>
#include <stdint.h>

#define N_ROWS 10000
#define DHID   256
#define CSPLIT 5
#define CT_PER 125   // 625 column tiles / CSPLIT

typedef __attribute__((ext_vector_type(16))) __bf16 v16bf;
typedef __attribute__((ext_vector_type(8)))  float  v8f;

__device__ __forceinline__ unsigned short f2bf(float x) {
  unsigned int u = __float_as_uint(x);
  u += 0x7FFFu + ((u >> 16) & 1u);          // round-to-nearest-even
  return (unsigned short)(u >> 16);
}
__device__ __forceinline__ float bf2f(unsigned short h) {
  return __uint_as_float(((unsigned int)h) << 16);
}

// ---------------------------------------------------------------------------
// Kernel A: h = elu(z @ W1^T + b1); out = h @ W2^T + b2; p = out / max(||out||,eps)
// ---------------------------------------------------------------------------
__global__ __launch_bounds__(256)
void proj_norm_kernel(const float* __restrict__ z1, const float* __restrict__ z2,
                      const float* __restrict__ z3,
                      const float* __restrict__ W1, const float* __restrict__ b1,
                      const float* __restrict__ W2, const float* __restrict__ b2,
                      unsigned short* __restrict__ p1,
                      unsigned short* __restrict__ p2,
                      unsigned short* __restrict__ p3) {
  __shared__ float zbuf[32][DHID];   // 32 KB
  __shared__ float hbuf[32][DHID];   // 32 KB (re-used for inv-norms at the end)
  const int tid   = threadIdx.x;
  const int which = blockIdx.y;
  const float* z = (which == 0) ? z1 : ((which == 1) ? z2 : z3);
  unsigned short* po = (which == 0) ? p1 : ((which == 1) ? p2 : p3);
  const int rbase = blockIdx.x * 32;

  for (int j = 0; j < 32; ++j) {
    int grow = rbase + j; if (grow > N_ROWS - 1) grow = N_ROWS - 1;
    zbuf[j][tid] = z[(size_t)grow * DHID + tid];
  }
  __syncthreads();

  const int c = tid;
  float acc[32];

  #pragma unroll
  for (int j = 0; j < 32; ++j) acc[j] = 0.f;
  {
    const float4* w4 = reinterpret_cast<const float4*>(W1 + (size_t)c * DHID);
    for (int k4 = 0; k4 < DHID / 4; ++k4) {
      float4 w = w4[k4];
      #pragma unroll
      for (int j = 0; j < 32; ++j) {
        float4 zz = reinterpret_cast<const float4*>(&zbuf[j][0])[k4];
        acc[j] = fmaf(w.x, zz.x, acc[j]);
        acc[j] = fmaf(w.y, zz.y, acc[j]);
        acc[j] = fmaf(w.z, zz.z, acc[j]);
        acc[j] = fmaf(w.w, zz.w, acc[j]);
      }
    }
  }
  {
    float bb = b1[c];
    #pragma unroll
    for (int j = 0; j < 32; ++j) {
      float x = acc[j] + bb;
      hbuf[j][c] = (x > 0.f) ? x : (__expf(x) - 1.f);   // ELU(alpha=1)
    }
  }
  __syncthreads();

  #pragma unroll
  for (int j = 0; j < 32; ++j) acc[j] = 0.f;
  {
    const float4* w4 = reinterpret_cast<const float4*>(W2 + (size_t)c * DHID);
    for (int k4 = 0; k4 < DHID / 4; ++k4) {
      float4 w = w4[k4];
      #pragma unroll
      for (int j = 0; j < 32; ++j) {
        float4 hh = reinterpret_cast<const float4*>(&hbuf[j][0])[k4];
        acc[j] = fmaf(w.x, hh.x, acc[j]);
        acc[j] = fmaf(w.y, hh.y, acc[j]);
        acc[j] = fmaf(w.z, hh.z, acc[j]);
        acc[j] = fmaf(w.w, hh.w, acc[j]);
      }
    }
  }
  {
    float bb = b2[c];
    #pragma unroll
    for (int j = 0; j < 32; ++j) zbuf[j][c] = acc[j] + bb;
  }
  __syncthreads();

  float* invn = &hbuf[0][0];
  if (tid < 32) {
    float s = 0.f;
    for (int k = 0; k < DHID; ++k) { float v = zbuf[tid][k]; s = fmaf(v, v, s); }
    float n = sqrtf(s);
    n = (n > 1e-12f) ? n : 1e-12f;
    invn[tid] = 1.f / n;
  }
  __syncthreads();

  #pragma unroll
  for (int j = 0; j < 32; ++j) {
    int grow = rbase + j;
    if (grow < N_ROWS)
      po[(size_t)grow * DHID + c] = f2bf(zbuf[j][c] * invn[j]);
  }
}

// ---------------------------------------------------------------------------
// Async global -> LDS staging of one 16-col x 256-K bf16 tile (8 KB).
// Each of 256 threads moves 32 B via two GLOBAL_LOAD_ASYNC_TO_LDS_B128.
// The instruction offset applies to both the global and LDS address (ISA 08).
// ---------------------------------------------------------------------------
__device__ __forceinline__ void async_tile_load(uint32_t lds_off,
                                                const unsigned short* gbase,
                                                int tid) {
  uint64_t g = (uint64_t)(uintptr_t)gbase + (uint32_t)tid * 16u;
  uint32_t l = lds_off + (uint32_t)tid * 16u;
  asm volatile("global_load_async_to_lds_b128 %0, %1, off"
               :: "v"(l), "v"(g) : "memory");
  asm volatile("global_load_async_to_lds_b128 %0, %1, off offset:4096"
               :: "v"(l), "v"(g) : "memory");
}

// ---------------------------------------------------------------------------
// Kernel B: fused exp-rowsum of the similarity matrices via bf16 WMMA.
// grid = (40 row blocks of 256 rows, CSPLIT column chunks, 6 pair slots)
// Each wave owns two 16-row strips -> two independent WMMA chains sharing
// every B fragment; B tiles double-buffered in LDS via async loads.
// slots: 0:S11 1:S12 2:S13 3:S22 4:S21 5:S23  (row sums of exp(2*dot))
// ---------------------------------------------------------------------------
__global__ __launch_bounds__(256)
void sim_rowsum_kernel(const unsigned short* __restrict__ p1,
                       const unsigned short* __restrict__ p2,
                       const unsigned short* __restrict__ p3,
                       float* __restrict__ rowsums) {
  __shared__ __align__(32) uint4 ldsq[2][512];   // 2 x 8 KB B tiles
  const int tid  = threadIdx.x;
  const int lane = tid & 31;
  const int wave = tid >> 5;
  const int rb = blockIdx.x;
  const int cs = blockIdx.y;
  const int p  = blockIdx.z;

  const unsigned short* pa = (p < 3) ? p1 : p2;
  const unsigned short* pb =
      (p == 0) ? p1 : (p == 1) ? p2 : (p == 2) ? p3 :
      (p == 3) ? p2 : (p == 4) ? p1 : p3;

  const int mr = rb * 256 + wave * 32;   // two 16-row strips per wave
  const int lm = lane & 15;
  const int kg = lane >> 4;

  // A fragments (contiguous K order, matching B): K = kb + 16*kg + i
  v16bf afragA[8], afragB[8];
  {
    int rowA = mr + lm;      if (rowA > N_ROWS - 1) rowA = N_ROWS - 1;
    int rowB = mr + 16 + lm; if (rowB > N_ROWS - 1) rowB = N_ROWS - 1;
    const unsigned short* arowA = pa + (size_t)rowA * DHID;
    const unsigned short* arowB = pa + (size_t)rowB * DHID;
    #pragma unroll
    for (int kf = 0; kf < 8; ++kf) {
      int base = kf * 32 + 16 * kg;
      union { v16bf v; uint4 q[2]; } u;
      u.q[0] = *reinterpret_cast<const uint4*>(arowA + base);
      u.q[1] = *reinterpret_cast<const uint4*>(arowA + base + 8);
      afragA[kf] = u.v;
      u.q[0] = *reinterpret_cast<const uint4*>(arowB + base);
      u.q[1] = *reinterpret_cast<const uint4*>(arowB + base + 8);
      afragB[kf] = u.v;
    }
  }

  float acc0[8], acc1[8];
  #pragma unroll
  for (int r = 0; r < 8; ++r) { acc0[r] = 0.f; acc1[r] = 0.f; }

  const uint32_t lds0 = (uint32_t)(uintptr_t)&ldsq[0][0];
  const unsigned short* ldsh0 = reinterpret_cast<const unsigned short*>(&ldsq[0][0]);
  const int bidx = lm * DHID + 16 * kg;   // B frag: lane n holds K chunk [kb+16*kg, +16)

  const int ct0 = cs * CT_PER;
  const int ctEnd = ct0 + CT_PER;

  // prologue: prefetch first tile into buffer 0
  async_tile_load(lds0, pb + (size_t)ct0 * 16 * DHID, tid);

  int buf = 0;
  for (int ct = ct0; ct < ctEnd; ++ct, buf ^= 1) {
    if (ct + 1 < ctEnd) {
      async_tile_load(lds0 + (uint32_t)(buf ^ 1) * 8192u,
                      pb + (size_t)(ct + 1) * 16 * DHID, tid);
      asm volatile("s_wait_asynccnt 0x2" ::: "memory"); // older 2 (current buf) done
    } else {
      asm volatile("s_wait_asynccnt 0x0" ::: "memory");
    }
    __syncthreads();   // current buffer visible to all waves

    const unsigned short* ldsh = ldsh0 + (size_t)buf * 4096;
    v8f cf0 = {}, cf1 = {};
    #pragma unroll
    for (int kf = 0; kf < 8; ++kf) {
      v16bf bfrag = *reinterpret_cast<const v16bf*>(ldsh + bidx + kf * 32);
      cf0 = __builtin_amdgcn_wmma_f32_16x16x32_bf16(false, afragA[kf], false, bfrag,
                                                    (short)0, cf0, false, false);
      cf1 = __builtin_amdgcn_wmma_f32_16x16x32_bf16(false, afragB[kf], false, bfrag,
                                                    (short)0, cf1, false, false);
    }
    #pragma unroll
    for (int r = 0; r < 8; ++r) {
      acc0[r] += __expf(2.0f * cf0[r]);   // 1/tau = 2
      acc1[r] += __expf(2.0f * cf1[r]);
    }
    __syncthreads();   // all waves done with this buffer before it is refilled
  }

  // reduce over the 16 lanes holding one output row; atomic-free slice writes
  float* rs = rowsums + ((size_t)p * CSPLIT + cs) * N_ROWS;
  #pragma unroll
  for (int r = 0; r < 8; ++r) {
    float v0 = acc0[r];
    v0 += __shfl_xor(v0, 1, 16);
    v0 += __shfl_xor(v0, 2, 16);
    v0 += __shfl_xor(v0, 4, 16);
    v0 += __shfl_xor(v0, 8, 16);
    float v1 = acc1[r];
    v1 += __shfl_xor(v1, 1, 16);
    v1 += __shfl_xor(v1, 2, 16);
    v1 += __shfl_xor(v1, 4, 16);
    v1 += __shfl_xor(v1, 8, 16);
    if (lm == 0) {
      int rowA = mr + r + 8 * kg;           // C-layout: VGPR r -> M = r + 8*half
      int rowB = rowA + 16;
      if (rowA < N_ROWS) rs[rowA] = v0;
      if (rowB < N_ROWS) rs[rowB] = v1;
    }
  }
}

// ---------------------------------------------------------------------------
// Kernel C: data-dependent diagonals d12, d13, d23
// ---------------------------------------------------------------------------
__global__ __launch_bounds__(256)
void diag_kernel(const unsigned short* __restrict__ p1,
                 const unsigned short* __restrict__ p2,
                 const unsigned short* __restrict__ p3,
                 float* __restrict__ d12, float* __restrict__ d13,
                 float* __restrict__ d23) {
  int i = blockIdx.x * 256 + threadIdx.x;
  if (i >= N_ROWS) return;
  const unsigned short* a = p1 + (size_t)i * DHID;
  const unsigned short* b = p2 + (size_t)i * DHID;
  const unsigned short* c = p3 + (size_t)i * DHID;
  float s12 = 0.f, s13 = 0.f, s23 = 0.f;
  for (int k = 0; k < DHID; ++k) {
    float x = bf2f(a[k]), y = bf2f(b[k]), z = bf2f(c[k]);
    s12 = fmaf(x, y, s12); s13 = fmaf(x, z, s13); s23 = fmaf(y, z, s23);
  }
  d12[i] = s12; d13[i] = s13; d23[i] = s23;
}

// ---------------------------------------------------------------------------
// Kernel D: assemble per-row losses, mean-reduce to scalar.
// ---------------------------------------------------------------------------
__global__ __launch_bounds__(256)
void loss_kernel(const float* __restrict__ rowsums,
                 const float* __restrict__ d12, const float* __restrict__ d13,
                 const float* __restrict__ d23, float* __restrict__ out) {
  __shared__ float red[256];
  const int tid = threadIdx.x;
  const float e2 = __expf(2.0f);
  float local = 0.f;
  for (int i = tid; i < N_ROWS; i += 256) {
    float rs[6];
    #pragma unroll
    for (int s = 0; s < 6; ++s) {
      float v = 0.f;
      #pragma unroll
      for (int cs = 0; cs < CSPLIT; ++cs)
        v += rowsums[((size_t)s * CSPLIT + cs) * N_ROWS + i];
      rs[s] = v;
    }
    float a12 = 2.f * d12[i], a13 = 2.f * d13[i], a23 = 2.f * d23[i];
    float eia = __expf(a12);
    float den1 = rs[0] + rs[1] + rs[2] - e2 - eia - __expf(a13);
    float den2 = rs[3] + rs[4] + rs[5] - e2 - eia - __expf(a23);
    local += 0.5f * ((__logf(den1) - a12) + (__logf(den2) - a12));
  }
  red[tid] = local;
  __syncthreads();
  for (int s = 128; s > 0; s >>= 1) {
    if (tid < s) red[tid] += red[tid + s];
    __syncthreads();
  }
  if (tid == 0) out[0] = red[0] / (float)N_ROWS;
}

// ---------------------------------------------------------------------------
extern "C" void kernel_launch(void* const* d_in, const int* in_sizes, int n_in,
                              void* d_out, int out_size, void* d_ws, size_t ws_size,
                              hipStream_t stream) {
  (void)in_sizes; (void)n_in; (void)out_size; (void)ws_size;
  const float* z1 = (const float*)d_in[0];
  const float* z2 = (const float*)d_in[1];
  const float* z3 = (const float*)d_in[2];
  const float* W1 = (const float*)d_in[3];
  const float* b1 = (const float*)d_in[4];
  const float* W2 = (const float*)d_in[5];
  const float* b2 = (const float*)d_in[6];

  char* ws = (char*)d_ws;
  const size_t PH = (size_t)N_ROWS * DHID * sizeof(unsigned short); // 5,120,000 B
  unsigned short* p1 = (unsigned short*)(ws);
  unsigned short* p2 = (unsigned short*)(ws + PH);
  unsigned short* p3 = (unsigned short*)(ws + 2 * PH);
  float* rowsums = (float*)(ws + 3 * PH);              // 6*CSPLIT*N floats = 1.2 MB
  float* d12 = rowsums + (size_t)6 * CSPLIT * N_ROWS;
  float* d13 = d12 + N_ROWS;
  float* d23 = d13 + N_ROWS;

  proj_norm_kernel<<<dim3(313, 3), 256, 0, stream>>>(z1, z2, z3, W1, b1, W2, b2,
                                                     p1, p2, p3);
  diag_kernel<<<(N_ROWS + 255) / 256, 256, 0, stream>>>(p1, p2, p3, d12, d13, d23);
  sim_rowsum_kernel<<<dim3(40, CSPLIT, 6), 256, 0, stream>>>(p1, p2, p3, rowsums);
  loss_kernel<<<1, 256, 0, stream>>>(rowsums, d12, d13, d23, (float*)d_out);
}
